// SelectiveScan_8572754723187
// MI455X (gfx1250) — matrix operand order repaired and checked by reading, not compiled, not run on hardware
//
#include <hip/hip_runtime.h>

// ---------------------------------------------------------------------------
// Problem constants (reference: B=2, L=2048, D=1024, N=16, K=4, R=64)
// ---------------------------------------------------------------------------
#define BB   2
#define LL   2048
#define DD   1024
#define NST  16
#define KC   4
#define RR   64
#define MM   (BB * LL)        // 4096 rows for all GEMMs
#define P_LD 128              // padded leading dim for params (96 -> 128)

typedef __attribute__((ext_vector_type(16))) __bf16 v16bf;
typedef __attribute__((ext_vector_type(8)))  __bf16 v8bf;
typedef __attribute__((ext_vector_type(8)))  float  v8f;

__device__ __forceinline__ unsigned short f2bf(float f) {
    unsigned int u = __float_as_uint(f);
    unsigned int r = u + 0x7FFFu + ((u >> 16) & 1u);   // round-to-nearest-even
    return (unsigned short)(r >> 16);
}

__device__ __forceinline__ float silu(float x) {
    return x * (1.0f / (1.0f + __expf(-x)));
}

// ---------------------------------------------------------------------------
// Prep kernels: f32 -> bf16 activation copy, weight transpose to K-major bf16
// ---------------------------------------------------------------------------
__global__ void k_f32_to_bf16(const float* __restrict__ src,
                              unsigned short* __restrict__ dst, int n) {
    int i = blockIdx.x * blockDim.x + threadIdx.x;
    if (i < n) dst[i] = f2bf(src[i]);
}

// W is (nout, Kdim) row-major f32; Wt is (Kdim, npad) row-major bf16, zero padded
__global__ void k_transpose_bf16(const float* __restrict__ W,
                                 unsigned short* __restrict__ Wt,
                                 int nout, int Kdim, int npad) {
    int i = blockIdx.x * blockDim.x + threadIdx.x;
    int total = Kdim * npad;
    if (i >= total) return;
    int k = i / npad, n = i - k * npad;
    Wt[i] = (n < nout) ? f2bf(W[(size_t)n * Kdim + k]) : (unsigned short)0;
}

// extract dt_unproj (cols 0..63 of padded params) as bf16 GEMM operand
__global__ void k_cvt_dtu(const float* __restrict__ params,
                          unsigned short* __restrict__ dst) {
    int i = blockIdx.x * blockDim.x + threadIdx.x;
    if (i >= MM * RR) return;
    int m = i >> 6, r = i & 63;
    dst[i] = f2bf(params[(size_t)m * P_LD + r]);
}

// ---------------------------------------------------------------------------
// WMMA bf16 GEMM:  C[M x Npad(f32)] = A[M x K(bf16)] * Bt[K x Npad(bf16)]
// Bt is the K-major (pre-transposed, zero-padded) weight.
// One wave produces a 16x64 strip: 1 A-fragment reused across 4 WMMA per k-step.
// ---------------------------------------------------------------------------
__global__ __launch_bounds__(128) void k_gemm_bf16(
    const unsigned short* __restrict__ A,
    const unsigned short* __restrict__ Bt,
    float* __restrict__ C,
    int M, int K, int Npad, int Nreal)
{
    const int lane   = threadIdx.x & 31;
    const int wib    = threadIdx.x >> 5;
    const int tilesN = Npad >> 6;                       // 64-wide strips
    const int gw     = blockIdx.x * 4 + wib;            // global wave id
    const int mt     = gw / tilesN;
    const int nt     = gw - mt * tilesN;
    if (mt * 16 >= M) return;                           // wave-uniform guard

    const int m  = mt * 16 + (lane & 15);               // A row for this lane
    const int kk = (lane >> 4) * 8;                     // K sub-chunk select
    const int nb = nt * 64;

    v8f acc0 = {}, acc1 = {}, acc2 = {}, acc3 = {};

    for (int k = 0; k < K; k += 32) {
        // ---- A fragment: lane holds row m, K = {k+kk..+7, k+kk+16..+23} ----
        const __bf16* pa = (const __bf16*)(A + (size_t)m * K + k + kk);
        v8bf alo = *(const v8bf*)(pa);
        v8bf ahi = *(const v8bf*)(pa + 16);
        v16bf afrag;
#pragma unroll
        for (int i = 0; i < 8; ++i) { afrag[i] = alo[i]; afrag[i + 8] = ahi[i]; }

        // ---- B fragments: lane holds column (lane&15), K = {k+kk..+7, k+kk+16..+23} ----
        const size_t brow = (size_t)(k + kk) * Npad + nb + (lane & 15);
#pragma unroll
        for (int j = 0; j < 4; ++j) {
            const __bf16* pb = (const __bf16*)(Bt + brow + j * 16);
            v16bf bfrag;
#pragma unroll
            for (int i = 0; i < 8; ++i) { bfrag[i] = pb[(size_t)i * Npad]; bfrag[i + 8] = pb[(size_t)(i + 16) * Npad]; }
            if (j == 0) acc0 = __builtin_amdgcn_wmma_f32_16x16x32_bf16(false, afrag, false, bfrag, (short)0, acc0, false, false);
            if (j == 1) acc1 = __builtin_amdgcn_wmma_f32_16x16x32_bf16(false, afrag, false, bfrag, (short)0, acc1, false, false);
            if (j == 2) acc2 = __builtin_amdgcn_wmma_f32_16x16x32_bf16(false, afrag, false, bfrag, (short)0, acc2, false, false);
            if (j == 3) acc3 = __builtin_amdgcn_wmma_f32_16x16x32_bf16(false, afrag, false, bfrag, (short)0, acc3, false, false);
        }
    }

    // ---- store: VGPR r -> row mt*16 + r + (lane>=16 ? 8 : 0), col lane&15 ----
    const int mrow  = mt * 16 + ((lane >> 4) << 3);
    const int ncol0 = nb + (lane & 15);
#pragma unroll
    for (int j = 0; j < 4; ++j) {
        const int nc = ncol0 + j * 16;
        if (nc >= Nreal) continue;
        const v8f acc = (j == 0) ? acc0 : (j == 1) ? acc1 : (j == 2) ? acc2 : acc3;
#pragma unroll
        for (int r = 0; r < 8; ++r)
            C[(size_t)(mrow + r) * Npad + nc] = acc[r];
    }
}

// ---------------------------------------------------------------------------
// dt = softplus(dtraw + b_dt)   (in place)
// ---------------------------------------------------------------------------
__global__ void k_softplus(float* __restrict__ dt, const float* __restrict__ b_dt) {
    int i = blockIdx.x * blockDim.x + threadIdx.x;
    if (i >= MM * DD) return;
    int d = i & (DD - 1);
    float v = dt[i] + b_dt[d];
    dt[i] = (v > 20.0f) ? v : log1pf(__expf(v));
}

// ---------------------------------------------------------------------------
// causal depthwise conv (K=4) over x_res, then SiLU  -> u
// ---------------------------------------------------------------------------
__global__ void k_conv_silu(const float* __restrict__ xres,
                            const float* __restrict__ cw,
                            float* __restrict__ u_out) {
    int i = blockIdx.x * blockDim.x + threadIdx.x;   // over B*L*D
    if (i >= MM * DD) return;
    int d = i & (DD - 1);
    int t = (i >> 10) & (LL - 1);
    const float w0 = cw[d * KC + 0], w1 = cw[d * KC + 1],
                w2 = cw[d * KC + 2], w3 = cw[d * KC + 3];
    float acc = xres[i] * w3;
    if (t >= 1) acc += xres[i - 1 * DD] * w2;
    if (t >= 2) acc += xres[i - 2 * DD] * w1;
    if (t >= 3) acc += xres[i - 3 * DD] * w0;
    u_out[i] = silu(acc);
}

// ---------------------------------------------------------------------------
// Selective scan. One lane per (b,d,n) state; 16-lane groups reduce C.h via
// wave32 shuffles. B/C projections staged in LDS per 128-step tile (shared by
// all channels of the batch handled by this block). Epilogue fuses
// (y + u*D) * silu(z) and writes bf16 rows for the output GEMM.
// ---------------------------------------------------------------------------
#define SCAN_TT 128
__global__ __launch_bounds__(256) void k_scan(
    const float* __restrict__ params,   // (M, P_LD): cols 64..79 = B, 80..95 = C
    const float* __restrict__ dt,       // (M, D)
    const float* __restrict__ u,        // (M, D)
    const float* __restrict__ z,        // (M, D)
    const float* __restrict__ A_log,    // (D, N)
    const float* __restrict__ Dp,       // (D)
    unsigned short* __restrict__ ybf)   // (M, D) bf16
{
    __shared__ float lds[SCAN_TT * 32];
    const int tid  = threadIdx.x;
    const int n    = tid & 15;
    const int pair = blockIdx.x * 16 + (tid >> 4);   // (b,d) pair
    const int b    = pair >> 10;                     // D = 1024
    const int d    = pair & (DD - 1);

    const float a      = -__expf(A_log[d * NST + n]);
    const float dparam = Dp[d];
    const float binv   = 1.0f / (a + 1e-10f);
    const bool  tiny   = fabsf(a) < 1e-8f;
    float h = 0.0f;
    const size_t rowbase = (size_t)b * LL;

    for (int t0 = 0; t0 < LL; t0 += SCAN_TT) {
        __syncthreads();
        for (int idx = tid; idx < SCAN_TT * 32; idx += 256) {
            int tl = idx >> 5, c = idx & 31;
            lds[idx] = params[(rowbase + t0 + tl) * P_LD + 64 + c];
        }
        __syncthreads();
        for (int tl = 0; tl < SCAN_TT; ++tl) {
            const size_t row = rowbase + t0 + tl;
            const float dtv = dt[row * DD + d];
            const float uv  = u[row * DD + d];
            const float At  = __expf(dtv * a);
            const float Btl = tiny ? dtv : (At - 1.0f) * binv;
            h = At * h + Btl * lds[tl * 32 + n] * uv;
            float part = lds[tl * 32 + 16 + n] * h;
            part += __shfl_xor(part, 8, 16);
            part += __shfl_xor(part, 4, 16);
            part += __shfl_xor(part, 2, 16);
            part += __shfl_xor(part, 1, 16);
            if (n == 0) {
                const float zv = z[row * DD + d];
                const float yv = (part + uv * dparam) * silu(zv);
                ybf[row * DD + d] = f2bf(yv);
            }
        }
    }
}

// ---------------------------------------------------------------------------
// host: orchestrate the pipeline on `stream`, sub-allocating from d_ws
// ---------------------------------------------------------------------------
extern "C" void kernel_launch(void* const* d_in, const int* in_sizes, int n_in,
                              void* d_out, int out_size, void* d_ws, size_t ws_size,
                              hipStream_t stream) {
    (void)in_sizes; (void)n_in; (void)out_size; (void)ws_size;
    const float* x        = (const float*)d_in[0];
    const float* W_x      = (const float*)d_in[1];
    const float* W_z      = (const float*)d_in[2];
    const float* W_params = (const float*)d_in[3];
    const float* conv_w   = (const float*)d_in[4];
    const float* W_dt     = (const float*)d_in[5];
    const float* b_dt     = (const float*)d_in[6];
    const float* A_log    = (const float*)d_in[7];
    const float* D_param  = (const float*)d_in[8];
    const float* W_out    = (const float*)d_in[9];
    float* out = (float*)d_out;

    char* ws = (char*)d_ws;
    size_t off = 0;
    auto alloc = [&](size_t bytes) -> void* {
        void* p = ws + off;
        off = (off + bytes + 255) & ~(size_t)255;
        return p;
    };
    unsigned short* xbf     = (unsigned short*)alloc((size_t)MM * DD * 2);
    unsigned short* wt_x    = (unsigned short*)alloc((size_t)DD * DD * 2);
    unsigned short* wt_z    = (unsigned short*)alloc((size_t)DD * DD * 2);
    unsigned short* wt_p    = (unsigned short*)alloc((size_t)DD * P_LD * 2);
    unsigned short* wt_dt   = (unsigned short*)alloc((size_t)RR * DD * 2);
    unsigned short* wt_out  = (unsigned short*)alloc((size_t)DD * DD * 2);
    float*          xres    = (float*)alloc((size_t)MM * DD * 4);
    float*          zbuf    = (float*)alloc((size_t)MM * DD * 4);
    float*          paramsb = (float*)alloc((size_t)MM * P_LD * 4);
    unsigned short* dtu     = (unsigned short*)alloc((size_t)MM * RR * 2);
    float*          dtbuf   = (float*)alloc((size_t)MM * DD * 4);
    float*          ubuf    = (float*)alloc((size_t)MM * DD * 4);
    unsigned short* ybf     = (unsigned short*)alloc((size_t)MM * DD * 2);

    const int T = 256;
    auto blocks = [&](int n) { return (n + T - 1) / T; };

    // activations & weights to bf16 (weights transposed to K-major, padded)
    k_f32_to_bf16<<<blocks(MM * DD), T, 0, stream>>>(x, xbf, MM * DD);
    k_transpose_bf16<<<blocks(DD * DD),   T, 0, stream>>>(W_x,      wt_x,  DD, DD, DD);
    k_transpose_bf16<<<blocks(DD * DD),   T, 0, stream>>>(W_z,      wt_z,  DD, DD, DD);
    k_transpose_bf16<<<blocks(DD * P_LD), T, 0, stream>>>(W_params, wt_p,  RR + 2 * NST, DD, P_LD);
    k_transpose_bf16<<<blocks(RR * DD),   T, 0, stream>>>(W_dt,     wt_dt, DD, RR, DD);
    k_transpose_bf16<<<blocks(DD * DD),   T, 0, stream>>>(W_out,    wt_out, DD, DD, DD);

    auto gemm = [&](const unsigned short* A, const unsigned short* Bt, float* C,
                    int M, int K, int Npad, int Nreal) {
        int waves = (M / 16) * (Npad / 64);
        k_gemm_bf16<<<(waves + 3) / 4, 128, 0, stream>>>(A, Bt, C, M, K, Npad, Nreal);
    };

    // input projections (WMMA)
    gemm(xbf, wt_x, xres,    MM, DD, DD,   DD);
    gemm(xbf, wt_z, zbuf,    MM, DD, DD,   DD);
    gemm(xbf, wt_p, paramsb, MM, DD, P_LD, RR + 2 * NST);

    // dt projection (WMMA, K=64) + softplus
    k_cvt_dtu<<<blocks(MM * RR), T, 0, stream>>>(paramsb, dtu);
    gemm(dtu, wt_dt, dtbuf, MM, RR, DD, DD);
    k_softplus<<<blocks(MM * DD), T, 0, stream>>>(dtbuf, b_dt);

    // depthwise causal conv + SiLU
    k_conv_silu<<<blocks(MM * DD), T, 0, stream>>>(xres, conv_w, ubuf);

    // selective scan (lane-per-state, fused gating epilogue, bf16 output rows)
    k_scan<<<(BB * DD) / 16, 256, 0, stream>>>(paramsb, dtbuf, ubuf, zbuf,
                                               A_log, D_param, ybf);

    // output projection (WMMA) -> d_out
    gemm(ybf, wt_out, out, MM, DD, DD, DD);
}
